// GraphBiasedMHA_652835029775
// MI455X (gfx1250) — compile-verified, matrix operand order checked by reading, not run on hardware
//
#include <hip/hip_runtime.h>
#include <hip/hip_bf16.h>
#include <math.h>

typedef __attribute__((ext_vector_type(16))) _Float16 v16h;
typedef __attribute__((ext_vector_type(8)))  _Float16 v8h;
typedef __attribute__((ext_vector_type(8)))  float    v8f;

union HFrag { v16h v; v8h h[2]; };

static constexpr int B_    = 2;
static constexpr int NSEQ  = 2048;
static constexpr int D_    = 1024;
static constexpr int H_    = 16;
static constexpr int DH    = 64;      // head dim
static constexpr int PDIM  = 256;
static constexpr int DHID  = 512;     // D/2
static constexpr int MROWS = B_ * NSEQ;   // 4096
static constexpr int NOUTQ = 3 * D_;      // 3072

// load one 16x32 f16 fragment slice for this lane; p already includes +8*g
__device__ __forceinline__ void ldfrag(HFrag& f, const _Float16* p) {
  f.h[0] = *(const v8h*)(p);
  f.h[1] = *(const v8h*)(p + 16);
}

__device__ __forceinline__ v8f wmma16(const HFrag& a, const HFrag& b, v8f c) {
  return __builtin_amdgcn_wmma_f32_16x16x32_f16(false, a.v, false, b.v,
                                                (short)0, c, false, false);
}

// ---------------------------------------------------------------- converts
__global__ void f32_to_f16_kernel(const float* __restrict__ src,
                                  _Float16* __restrict__ dst, int n) {
  int i = blockIdx.x * blockDim.x + threadIdx.x;
  int stride = gridDim.x * blockDim.x;
  for (; i < n; i += stride) dst[i] = (_Float16)src[i];
}

// ------------------------------------------------- gate + tail of col_bias
__global__ void gate_tail_kernel(const float* __restrict__ trust,
                                 const float* __restrict__ tscale,
                                 float* __restrict__ gate,
                                 float* __restrict__ bias,
                                 int n_ctx, int n) {
  int i = blockIdx.x * blockDim.x + threadIdx.x;
  if (i >= n) return;
  if (i < n_ctx) {
    gate[i] = 1.f / (1.f + __expf(-tscale[0] * trust[i]));
  } else {
    gate[i] = 1.f;
    bias[i] = 0.f;
  }
}

// ---------------------------------- per-context-row bias MLP (tiny, VALU)
__global__ __launch_bounds__(256) void ctx_bias_kernel(
    const float* __restrict__ delta, const float* __restrict__ Wd1,
    const float* __restrict__ bd1,   const float* __restrict__ Wd2,
    const float* __restrict__ bd2,   const float* __restrict__ ppr,
    const float* __restrict__ alpha, float* __restrict__ bias) {
  int row = blockIdx.x;
  const float* drow = delta + (size_t)row * PDIM;
  float local = 0.f;
  for (int j = threadIdx.x; j < DHID; j += blockDim.x) {
    const float* wrow = Wd1 + (size_t)j * PDIM;
    float a = bd1[j];
    for (int p = 0; p < PDIM; ++p) a = fmaf(drow[p], wrow[p], a);
    float ge = 0.5f * a * (1.f + erff(a * 0.70710678118654752f));  // exact gelu
    local += ge * Wd2[j];
  }
  __shared__ float red[256];
  red[threadIdx.x] = local;
  __syncthreads();
  for (int s = 128; s > 0; s >>= 1) {
    if ((int)threadIdx.x < s) red[threadIdx.x] += red[threadIdx.x + s];
    __syncthreads();
  }
  if (threadIdx.x == 0) {
    float pv = ppr[row];
    pv = pv < 1e-8f ? 1e-8f : pv;
    bias[row] = alpha[0] * logf(pv) + red[0] + bd2[0];
  }
}

// ------------------------------------------------------- QKV WMMA GEMM
// out = xh[4096,1024] @ W_in^T + b_in, routed to q/k/v head buffers.
// q scaled by d^-0.5, v gated + stored transposed [B,H,d,N].
// Register double-buffered over k so loads overlap WMMA.
__global__ __launch_bounds__(256, 1) void qkv_gemm_kernel(
    const _Float16* __restrict__ xh, const _Float16* __restrict__ winh,
    const float* __restrict__ b_in,  const float* __restrict__ gate,
    _Float16* __restrict__ qh, _Float16* __restrict__ kh,
    _Float16* __restrict__ vth) {
  const int lane = threadIdx.x & 31;
  const int wave = threadIdx.x >> 5;
  const int l16  = lane & 15;
  const int g    = lane >> 4;
  const int m0 = blockIdx.x * 64 + (wave & 3) * 16;
  const int n0 = blockIdx.y * 128 + (wave >> 2) * 64;

  const _Float16* arow = xh + (size_t)(m0 + l16) * D_ + 8 * g;
  const _Float16* brow[4];
#pragma unroll
  for (int f = 0; f < 4; ++f)
    brow[f] = winh + (size_t)(n0 + f * 16 + l16) * D_ + 8 * g;

  v8f zero = {0.f, 0.f, 0.f, 0.f, 0.f, 0.f, 0.f, 0.f};
  v8f acc[4];
#pragma unroll
  for (int f = 0; f < 4; ++f) acc[f] = zero;

  HFrag a0, a1, b0[4], b1[4];
  ldfrag(a0, arow);
#pragma unroll
  for (int f = 0; f < 4; ++f) ldfrag(b0[f], brow[f]);

  for (int k0 = 0; k0 < D_ - 64; k0 += 64) {
    ldfrag(a1, arow + k0 + 32);
#pragma unroll
    for (int f = 0; f < 4; ++f) ldfrag(b1[f], brow[f] + k0 + 32);
#pragma unroll
    for (int f = 0; f < 4; ++f) acc[f] = wmma16(a0, b0[f], acc[f]);
    ldfrag(a0, arow + k0 + 64);
#pragma unroll
    for (int f = 0; f < 4; ++f) ldfrag(b0[f], brow[f] + k0 + 64);
#pragma unroll
    for (int f = 0; f < 4; ++f) acc[f] = wmma16(a1, b1[f], acc[f]);
  }
  ldfrag(a1, arow + D_ - 32);
#pragma unroll
  for (int f = 0; f < 4; ++f) ldfrag(b1[f], brow[f] + D_ - 32);
#pragma unroll
  for (int f = 0; f < 4; ++f) acc[f] = wmma16(a0, b0[f], acc[f]);
#pragma unroll
  for (int f = 0; f < 4; ++f) acc[f] = wmma16(a1, b1[f], acc[f]);

  const float qscale = 0.125f;  // 1/sqrt(64)
#pragma unroll
  for (int f = 0; f < 4; ++f) {
    int nn = n0 + f * 16 + l16;
    int sect = nn / D_;
    int jj = nn % D_;
    int h = jj / DH, c = jj % DH;
#pragma unroll
    for (int r = 0; r < 8; ++r) {
      int m = m0 + r + 8 * g;
      int bi = m / NSEQ, tok = m % NSEQ;
      float val = acc[f][r] + b_in[nn];
      size_t hb = (size_t)(bi * H_ + h);
      if (sect == 0)
        qh[(hb * NSEQ + tok) * DH + c] = (_Float16)(val * qscale);
      else if (sect == 1)
        kh[(hb * NSEQ + tok) * DH + c] = (_Float16)val;
      else
        vth[(hb * DH + c) * NSEQ + tok] = (_Float16)(val * gate[tok]);
    }
  }
}

// --------------------------------------------------- flash attention
// 1 wave per 16-query tile; 32-key chunks; online softmax over all 32
// lanes; K and V fragments loaded together at chunk top so V latency is
// hidden behind S-WMMAs + softmax.
__global__ __launch_bounds__(128, 1) void flash_attn_kernel(
    const _Float16* __restrict__ qh, const _Float16* __restrict__ kh,
    const _Float16* __restrict__ vth, const float* __restrict__ bias,
    _Float16* __restrict__ attn) {
  const int lane = threadIdx.x & 31;
  const int wave = threadIdx.x >> 5;
  const int l16  = lane & 15;
  const int g    = lane >> 4;

  const int tile = blockIdx.x * 4 + wave;
  const int qt = tile % (NSEQ / 16);
  const int bh = tile / (NSEQ / 16);

  __shared__ __align__(16) float    Ssh[4][16 * 32];
  __shared__ __align__(16) _Float16 Psh[4][16 * 32];
  __shared__ float rmax[4][16], rsum[4][16], rfac[4][16];
  __shared__ float hm[4][32], hs[4][32];

  float*    S  = Ssh[wave];
  _Float16* Pp = Psh[wave];

  const _Float16* qbase = qh + ((size_t)bh * NSEQ + (size_t)qt * 16) * DH;
  const _Float16* kball = kh + (size_t)bh * NSEQ * DH;
  const _Float16* vbase = vth + (size_t)bh * DH * NSEQ;

  HFrag aq[2];
  {
    const _Float16* qrow = qbase + (size_t)l16 * DH + 8 * g;
    ldfrag(aq[0], qrow);
    ldfrag(aq[1], qrow + 32);
  }

  v8f zero = {0.f, 0.f, 0.f, 0.f, 0.f, 0.f, 0.f, 0.f};
  v8f acc[4];
#pragma unroll
  for (int f = 0; f < 4; ++f) acc[f] = zero;

  if (lane < 16) {
    rmax[wave][lane] = -INFINITY;
    rsum[wave][lane] = 0.f;
  }
  __syncthreads();

  const int row  = lane & 15;   // softmax: 2 lanes per query row
  const int half = lane >> 4;

  for (int kc = 0; kc < NSEQ; kc += 32) {
    // ---- issue all global loads for this chunk up front
    HFrag bk[4];   // [keysub(2)][dchunk(2)]
    {
      const _Float16* krow  = kball + (size_t)(kc + l16) * DH + 8 * g;
      const _Float16* krow2 = krow + 16 * DH;
      ldfrag(bk[0], krow);        // keys 0-15, d 0-31
      ldfrag(bk[1], krow + 32);   // keys 0-15, d 32-63
      ldfrag(bk[2], krow2);       // keys 16-31, d 0-31
      ldfrag(bk[3], krow2 + 32);  // keys 16-31, d 32-63
    }
    HFrag bv[4];
#pragma unroll
    for (int f = 0; f < 4; ++f) {
      const _Float16* vrow = vbase + (size_t)(f * 16 + l16) * NSEQ + kc + 8 * g;
      ldfrag(bv[f], vrow);
    }

    // ---- S = q @ k^T (scale pre-folded into q)
    v8f s0 = zero, s1 = zero;
    s0 = wmma16(aq[0], bk[0], s0);
    s0 = wmma16(aq[1], bk[1], s0);
    s1 = wmma16(aq[0], bk[2], s1);
    s1 = wmma16(aq[1], bk[3], s1);

#pragma unroll
    for (int r = 0; r < 8; ++r) {
      S[(r + 8 * g) * 32 + l16]      = s0[r];
      S[(r + 8 * g) * 32 + 16 + l16] = s1[r];
    }
    __syncthreads();

    // ---- online softmax, all 32 lanes (each handles 16 of a row)
    {
      float* srow = S + row * 32 + half * 16;
      const float* brow = bias + kc + half * 16;
      float lmax = -INFINITY;
#pragma unroll
      for (int j = 0; j < 16; ++j) {
        float sv = srow[j] + brow[j];
        srow[j] = sv;
        lmax = fmaxf(lmax, sv);
      }
      hm[wave][lane] = lmax;
      __syncthreads();
      float mold = rmax[wave][row];
      float mx = fmaxf(mold, fmaxf(hm[wave][row], hm[wave][row + 16]));
      float lsum = 0.f;
      _Float16* prow = Pp + row * 32 + half * 16;
#pragma unroll
      for (int j = 0; j < 16; ++j) {
        float p = __expf(srow[j] - mx);
        prow[j] = (_Float16)p;
        lsum += p;
      }
      hs[wave][lane] = lsum;
      __syncthreads();
      if (half == 0) {
        float fr = __expf(mold - mx);
        rsum[wave][row] = rsum[wave][row] * fr + hs[wave][row] + hs[wave][row + 16];
        rmax[wave][row] = mx;
        rfac[wave][row] = fr;
      }
    }
    __syncthreads();

    // ---- rescale accumulator + P@V (V already in registers)
    float fr[8];
#pragma unroll
    for (int r = 0; r < 8; ++r) fr[r] = rfac[wave][r + 8 * g];
    HFrag ap;
    {
      const _Float16* prow = Pp + l16 * 32 + 8 * g;
      ap.h[0] = *(const v8h*)(prow);
      ap.h[1] = *(const v8h*)(prow + 16);
    }
#pragma unroll
    for (int f = 0; f < 4; ++f) {
#pragma unroll
      for (int r = 0; r < 8; ++r) acc[f][r] *= fr[r];
      acc[f] = wmma16(ap, bv[f], acc[f]);
    }
    __syncthreads();
  }

  if (lane < 16) rfac[wave][lane] = 1.f / rsum[wave][lane];
  __syncthreads();

  const int bi = bh / H_;
  const int h  = bh % H_;
#pragma unroll
  for (int f = 0; f < 4; ++f) {
#pragma unroll
    for (int r = 0; r < 8; ++r) {
      int tok = qt * 16 + r + 8 * g;
      float inv = rfac[wave][r + 8 * g];
      attn[((size_t)(bi * NSEQ + tok)) * D_ + h * DH + f * 16 + l16] =
          (_Float16)(acc[f][r] * inv);
    }
  }
}

// ------------------------------------------------- output projection
__global__ __launch_bounds__(256, 1) void out_gemm_kernel(
    const _Float16* __restrict__ ah, const _Float16* __restrict__ wouth,
    const float* __restrict__ b_out, float* __restrict__ out) {
  const int lane = threadIdx.x & 31;
  const int wave = threadIdx.x >> 5;
  const int l16  = lane & 15;
  const int g    = lane >> 4;
  const int m0 = blockIdx.x * 64 + (wave & 3) * 16;
  const int n0 = blockIdx.y * 128 + (wave >> 2) * 64;

  const _Float16* arow = ah + (size_t)(m0 + l16) * D_ + 8 * g;
  const _Float16* brow[4];
#pragma unroll
  for (int f = 0; f < 4; ++f)
    brow[f] = wouth + (size_t)(n0 + f * 16 + l16) * D_ + 8 * g;

  v8f zero = {0.f, 0.f, 0.f, 0.f, 0.f, 0.f, 0.f, 0.f};
  v8f acc[4];
#pragma unroll
  for (int f = 0; f < 4; ++f) acc[f] = zero;

  HFrag a0, a1, b0[4], b1[4];
  ldfrag(a0, arow);
#pragma unroll
  for (int f = 0; f < 4; ++f) ldfrag(b0[f], brow[f]);

  for (int k0 = 0; k0 < D_ - 64; k0 += 64) {
    ldfrag(a1, arow + k0 + 32);
#pragma unroll
    for (int f = 0; f < 4; ++f) ldfrag(b1[f], brow[f] + k0 + 32);
#pragma unroll
    for (int f = 0; f < 4; ++f) acc[f] = wmma16(a0, b0[f], acc[f]);
    ldfrag(a0, arow + k0 + 64);
#pragma unroll
    for (int f = 0; f < 4; ++f) ldfrag(b0[f], brow[f] + k0 + 64);
#pragma unroll
    for (int f = 0; f < 4; ++f) acc[f] = wmma16(a1, b1[f], acc[f]);
  }
  ldfrag(a1, arow + D_ - 32);
#pragma unroll
  for (int f = 0; f < 4; ++f) ldfrag(b1[f], brow[f] + D_ - 32);
#pragma unroll
  for (int f = 0; f < 4; ++f) acc[f] = wmma16(a0, b0[f], acc[f]);
#pragma unroll
  for (int f = 0; f < 4; ++f) acc[f] = wmma16(a1, b1[f], acc[f]);

#pragma unroll
  for (int f = 0; f < 4; ++f) {
    int nn = n0 + f * 16 + l16;
#pragma unroll
    for (int r = 0; r < 8; ++r) {
      int m = m0 + r + 8 * g;
      out[(size_t)m * D_ + nn] = acc[f][r] + b_out[nn];
    }
  }
}

// ------------------------------------------------------------ launcher
extern "C" void kernel_launch(void* const* d_in, const int* in_sizes, int n_in,
                              void* d_out, int out_size, void* d_ws,
                              size_t ws_size, hipStream_t stream) {
  (void)n_in; (void)out_size; (void)ws_size;
  const float* x         = (const float*)d_in[0];
  const float* ctx_ppr   = (const float*)d_in[2];
  const float* ctx_delta = (const float*)d_in[3];
  const float* ctx_trust = (const float*)d_in[4];
  const float* W_in      = (const float*)d_in[5];
  const float* b_in      = (const float*)d_in[6];
  const float* W_out     = (const float*)d_in[7];
  const float* b_out     = (const float*)d_in[8];
  const float* alpha     = (const float*)d_in[9];
  const float* tscale    = (const float*)d_in[10];
  const float* Wd1       = (const float*)d_in[11];
  const float* bd1       = (const float*)d_in[12];
  const float* Wd2       = (const float*)d_in[13];
  const float* bd2       = (const float*)d_in[14];
  float* out = (float*)d_out;
  const int n_ctx = in_sizes[2];  // ctx_ppr has n_ctx elements

  // workspace carve (~42 MB; attn output reuses xh)
  char* w = (char*)d_ws;
  auto carve = [&](size_t bytes) -> char* {
    char* p = w;
    w += (bytes + 255) & ~(size_t)255;
    return p;
  };
  _Float16* xh    = (_Float16*)carve((size_t)MROWS * D_ * 2);
  _Float16* winh  = (_Float16*)carve((size_t)3 * D_ * D_ * 2);
  _Float16* wouth = (_Float16*)carve((size_t)D_ * D_ * 2);
  _Float16* qh    = (_Float16*)carve((size_t)MROWS * D_ * 2);
  _Float16* kh    = (_Float16*)carve((size_t)MROWS * D_ * 2);
  _Float16* vth   = (_Float16*)carve((size_t)MROWS * D_ * 2);
  float* gate = (float*)carve((size_t)NSEQ * 4);
  float* bias = (float*)carve((size_t)NSEQ * 4);
  _Float16* attn = xh;  // xh dead after QKV GEMM

  f32_to_f16_kernel<<<512, 256, 0, stream>>>(x, xh, MROWS * D_);
  f32_to_f16_kernel<<<512, 256, 0, stream>>>(W_in, winh, 3 * D_ * D_);
  f32_to_f16_kernel<<<256, 256, 0, stream>>>(W_out, wouth, D_ * D_);

  gate_tail_kernel<<<(NSEQ + 255) / 256, 256, 0, stream>>>(
      ctx_trust, tscale, gate, bias, n_ctx, NSEQ);
  ctx_bias_kernel<<<n_ctx, 256, 0, stream>>>(ctx_delta, Wd1, bd1, Wd2, bd2,
                                             ctx_ppr, alpha, bias);

  qkv_gemm_kernel<<<dim3(MROWS / 64, NOUTQ / 128), 256, 0, stream>>>(
      xh, winh, b_in, gate, qh, kh, vth);

  flash_attn_kernel<<<(B_ * H_ * (NSEQ / 16)) / 4, 128, 0, stream>>>(
      qh, kh, vth, bias, attn);

  out_gemm_kernel<<<dim3(MROWS / 64, D_ / 128), 256, 0, stream>>>(
      attn, wouth, b_out, out);
}
